// GraphEncoder_88923002896566
// MI455X (gfx1250) — matrix operand order repaired and checked by reading, not compiled, not run on hardware
//
#include <hip/hip_runtime.h>
#include <hip/hip_bf16.h>

typedef _Float16 half_t;
typedef __attribute__((ext_vector_type(16))) _Float16 v16h;
typedef __attribute__((ext_vector_type(8)))  _Float16 v8h;
typedef __attribute__((ext_vector_type(8)))  float    v8f;

#define D_MODEL 512
#define SEQL    128
#define NIMG    128
#define NTOT    8192
#define NHEAD   8
#define NLAYER  4
#define DFFN    2048
#define ATTRW   35
#define SIZEB   10
#define NLOC    25

static __device__ __forceinline__ v8f wmma16x16x32(v16h a, v16h b, v8f c) {
  // D = A(16x32 f16) * B(32x16 f16) + C(16x16 f32)
  return __builtin_amdgcn_wmma_f32_16x16x32_f16(false, a, false, b, (short)0, c, false, false);
}

// ---------------------------------------------------------------- setup kernels

__global__ __launch_bounds__(128) void zero_counts_kernel(int* counts) {
  if (threadIdx.x < NIMG) counts[threadIdx.x] = 0;
}

__global__ __launch_bounds__(256) void count_objs_kernel(const int* __restrict__ obj_to_img,
                                                         int* counts) {
  int n = blockIdx.x * 256 + threadIdx.x;
  if (n < NTOT) atomicAdd(&counts[obj_to_img[n]], 1);
}

__global__ __launch_bounds__(32) void scan_offsets_kernel(const int* __restrict__ counts,
                                                          int* offsets) {
  if (threadIdx.x == 0) {
    int acc = 0;
    for (int i = 0; i < NIMG; ++i) { offsets[i] = acc; acc += counts[i]; }
  }
}

__global__ __launch_bounds__(256) void write_masks_kernel(const int* __restrict__ counts,
                                                          float* __restrict__ masks) {
  int idx = blockIdx.x * 256 + threadIdx.x;   // 128*128
  if (idx < NIMG * SEQL) {
    int b = idx >> 7, s = idx & 127;
    masks[idx] = (s >= counts[b] + 1) ? 1.0f : 0.0f;
  }
}

__global__ __launch_bounds__(256) void f32_to_f16_kernel(const float* __restrict__ src,
                                                         half_t* __restrict__ dst, int n) {
  int i = blockIdx.x * 256 + threadIdx.x;
  if (i < n) dst[i] = (_Float16)src[i];
}

// h rows: CLS token (s==0) and zero fill elsewhere; writes f32 + f16 mirrors
__global__ __launch_bounds__(128) void init_h_kernel(const float* __restrict__ emb_token,
                                                     float* __restrict__ h,
                                                     half_t* __restrict__ hh) {
  int r = blockIdx.x;              // 16384 rows
  int s = r & (SEQL - 1);
  const float rate511 = powf(10000.0f, -510.0f / 512.0f);
  float*  dst  = h  + (size_t)r * D_MODEL;
  half_t* dsth = hh + (size_t)r * D_MODEL;
#pragma unroll
  for (int t = 0; t < 4; ++t) {
    int j = threadIdx.x + 128 * t;
    float val = 0.0f;
    if (s == 0) val = emb_token[j] + cosf((float)j * rate511);
    dst[j] = val;
    dsth[j] = (_Float16)val;
  }
}

// object rows: obj_vecs*sqrt(D) + positional_encoding[loc_idx]
__global__ __launch_bounds__(128) void fill_objects_kernel(const float* __restrict__ obj_vecs,
                                                           const float* __restrict__ attr,
                                                           const int* __restrict__ obj_to_img,
                                                           const int* __restrict__ offsets,
                                                           float* __restrict__ h,
                                                           half_t* __restrict__ hh) {
  __shared__ int sloc;
  int n = blockIdx.x;              // 8192 objects
  if (threadIdx.x == 0) {
    int best = 0; float bv = attr[(size_t)n * ATTRW + SIZEB];
    for (int c = 1; c < NLOC; ++c) {
      float v = attr[(size_t)n * ATTRW + SIZEB + c];
      if (v > bv) { bv = v; best = c; }
    }
    sloc = best;
  }
  __syncthreads();
  int loc = sloc;
  int img = obj_to_img[n];
  int pos = n - offsets[img];
  size_t rowoff = (size_t)(img * SEQL + pos + 1) * D_MODEL;
  float*  dst  = h  + rowoff;
  half_t* dsth = hh + rowoff;
  const float* src = obj_vecs + (size_t)n * D_MODEL;
  const float scale = sqrtf((float)D_MODEL);
#pragma unroll
  for (int t = 0; t < 4; ++t) {
    int j = threadIdx.x + 128 * t;
    float rate = powf(10000.0f, -(float)(2 * (j / 2)) / (float)D_MODEL);
    float ang = (float)loc * rate;
    float pe = (j & 1) ? cosf(ang) : sinf(ang);
    float val = src[j] * scale + pe;
    dst[j] = val;
    dsth[j] = (_Float16)val;
  }
}

// ---------------------------------------------------------------- WMMA GEMM (all-f16 inputs)
// C[M,N] = act( A[M,K] @ W[K,N] + bias[N] (+ resid[M,N]) )
// block = 128 threads (4 waves); block tile 64x64; wave tile 32x32 (2x2 WMMA)
template <typename OT, bool RELU, bool RESID>
__global__ __launch_bounds__(128) void gemm_wmma_kernel(const half_t* __restrict__ A, int lda,
                                                        const half_t* __restrict__ W, int ldb,
                                                        const float* __restrict__ bias,
                                                        const float* __restrict__ resid,
                                                        OT* __restrict__ C, int ldc, int K) {
  const int lane = threadIdx.x & 31;
  const int wave = threadIdx.x >> 5;
  const int g = lane >> 4;          // half-wave group
  const int mn = lane & 15;
  const int rowBase = blockIdx.y * 64 + (wave >> 1) * 32;
  const int colBase = blockIdx.x * 64 + (wave & 1) * 32;

  v8f c00 = {}, c01 = {}, c10 = {}, c11 = {};

  for (int kk = 0; kk < K; kk += 32) {
    v16h a0, a1, b0, b1;
    const half_t* arow0 = A + (size_t)(rowBase + mn) * lda + kk + g * 8;
    const half_t* arow1 = A + (size_t)(rowBase + 16 + mn) * lda + kk + g * 8;
    v8h a0lo = *(const v8h*)arow0;
    v8h a0hi = *(const v8h*)(arow0 + 16);
    v8h a1lo = *(const v8h*)arow1;
    v8h a1hi = *(const v8h*)(arow1 + 16);
#pragma unroll
    for (int j = 0; j < 8; ++j) {
      a0[j] = a0lo[j]; a0[8 + j] = a0hi[j];
      a1[j] = a1lo[j]; a1[8 + j] = a1hi[j];
    }
    const int kb = kk + g * 8;
#pragma unroll
    for (int j = 0; j < 16; ++j) {
      const half_t* wrow = W + (size_t)(kb + j + (j >> 3) * 8) * ldb + colBase;
      b0[j] = wrow[mn];
      b1[j] = wrow[16 + mn];
    }
    c00 = wmma16x16x32(a0, b0, c00);
    c01 = wmma16x16x32(a0, b1, c01);
    c10 = wmma16x16x32(a1, b0, c10);
    c11 = wmma16x16x32(a1, b1, c11);
  }

#pragma unroll
  for (int tt = 0; tt < 4; ++tt) {
    v8f acc = (tt == 0) ? c00 : (tt == 1) ? c01 : (tt == 2) ? c10 : c11;
    int rt = rowBase + (tt >> 1) * 16;
    int ct = colBase + (tt & 1) * 16;
    int col = ct + mn;
    float bv = bias[col];
#pragma unroll
    for (int i = 0; i < 8; ++i) {
      int row = rt + 8 * g + i;
      float val = acc[i] + bv;
      if constexpr (RESID) val += resid[(size_t)row * ldc + col];
      if constexpr (RELU)  val = fmaxf(val, 0.0f);
      C[(size_t)row * ldc + col] = (OT)val;
    }
  }
}

// ---------------------------------------------------------------- attention
// one block per (image, head); 256 threads = 8 waves
// dynamic LDS: 64KB f32 scores + 32KB f16 probs
__global__ __launch_bounds__(256) void attn_kernel(const half_t* __restrict__ q,
                                                   const half_t* __restrict__ k,
                                                   const half_t* __restrict__ v,
                                                   const float* __restrict__ masks,
                                                   float* __restrict__ atts,   // layer slice
                                                   half_t* __restrict__ o) {
  extern __shared__ char smem[];
  float  (*S)[SEQL] = (float(*)[SEQL])smem;                     // 64 KB
  half_t (*P)[SEQL] = (half_t(*)[SEQL])(smem + 64 * 1024);      // 32 KB
  const int bh = blockIdx.x;
  const int b = bh >> 3, hh = bh & 7;
  const int lane = threadIdx.x & 31;
  const int wave = threadIdx.x >> 5;
  const int g = lane >> 4, mn = lane & 15;
  const int qb = wave * 16;         // 16 query rows per wave
  const int hd0 = hh * 64;

  // ---- scores = Q K^T / 8 + mask * -1e9
  v8f acc[8] = {};
  for (int kk = 0; kk < 64; kk += 32) {
    v16h a;
    const half_t* ap = q + (size_t)(b * SEQL + qb + mn) * D_MODEL + hd0 + kk + g * 8;
    v8h alo = *(const v8h*)ap;
    v8h ahi = *(const v8h*)(ap + 16);
#pragma unroll
    for (int j = 0; j < 8; ++j) { a[j] = alo[j]; a[8 + j] = ahi[j]; }
#pragma unroll
    for (int t = 0; t < 8; ++t) {
      v16h bb;
      const half_t* kp = k + (size_t)(b * SEQL + t * 16 + mn) * D_MODEL + hd0 + kk + g * 8;
      v8h klo = *(const v8h*)kp;
      v8h khi = *(const v8h*)(kp + 16);
#pragma unroll
      for (int j = 0; j < 8; ++j) { bb[j] = klo[j]; bb[8 + j] = khi[j]; }
      acc[t] = wmma16x16x32(a, bb, acc[t]);
    }
  }
#pragma unroll
  for (int t = 0; t < 8; ++t) {
    int col = t * 16 + mn;
    float m = masks[b * SEQL + col];
#pragma unroll
    for (int i = 0; i < 8; ++i)
      S[qb + 8 * g + i][col] = acc[t][i] * 0.125f + m * -1000000000.0f;
  }
  __syncthreads();

  // ---- row softmax; probs -> f16 LDS and f32 atts output
  if (threadIdx.x < SEQL) {
    int r = threadIdx.x;
    float mx = -INFINITY;
    for (int c = 0; c < SEQL; ++c) mx = fmaxf(mx, S[r][c]);
    float sum = 0.0f;
    for (int c = 0; c < SEQL; ++c) { float e = expf(S[r][c] - mx); S[r][c] = e; sum += e; }
    float inv = 1.0f / sum;
    float* arow = atts + (((size_t)(b * NHEAD + hh)) * SEQL + r) * SEQL;
    for (int c = 0; c < SEQL; ++c) {
      float p = S[r][c] * inv;
      P[r][c] = (_Float16)p;
      arow[c] = p;
    }
  }
  __syncthreads();

  // ---- O = P @ V  (16 rows x 64 cols per wave)
  v8f oc[4] = {};
  for (int kk = 0; kk < SEQL; kk += 32) {
    v16h a;
    const half_t* pp = &P[qb + mn][kk + g * 8];
    v8h plo = *(const v8h*)pp;
    v8h phi = *(const v8h*)(pp + 16);
#pragma unroll
    for (int j = 0; j < 8; ++j) { a[j] = plo[j]; a[8 + j] = phi[j]; }
#pragma unroll
    for (int t = 0; t < 4; ++t) {
      v16h bb;
      const half_t* vp = v + (size_t)(b * SEQL + kk + g * 8) * D_MODEL + hd0 + t * 16 + mn;
#pragma unroll
      for (int j = 0; j < 16; ++j) bb[j] = vp[(size_t)(j + (j >> 3) * 8) * D_MODEL];
      oc[t] = wmma16x16x32(a, bb, oc[t]);
    }
  }
#pragma unroll
  for (int t = 0; t < 4; ++t) {
    int col = hd0 + t * 16 + mn;
#pragma unroll
    for (int i = 0; i < 8; ++i) {
      int row = b * SEQL + qb + 8 * g + i;
      o[(size_t)row * D_MODEL + col] = (_Float16)oc[t][i];
    }
  }
}

// ---------------------------------------------------------------- layernorm (writes f32 + f16)
__global__ __launch_bounds__(128) void ln_kernel(const float* __restrict__ y,
                                                 const float* __restrict__ gamma,
                                                 const float* __restrict__ beta,
                                                 float* __restrict__ hout,
                                                 half_t* __restrict__ hhout) {
  __shared__ float red[128];
  int r = blockIdx.x;
  const float* yr = y + (size_t)r * D_MODEL;
  float local[4];
  float s = 0.0f;
#pragma unroll
  for (int t = 0; t < 4; ++t) { local[t] = yr[threadIdx.x + 128 * t]; s += local[t]; }
  red[threadIdx.x] = s;
  __syncthreads();
  for (int st = 64; st > 0; st >>= 1) {
    if (threadIdx.x < st) red[threadIdx.x] += red[threadIdx.x + st];
    __syncthreads();
  }
  float mean = red[0] / (float)D_MODEL;
  __syncthreads();
  float vs = 0.0f;
#pragma unroll
  for (int t = 0; t < 4; ++t) { float d = local[t] - mean; vs += d * d; }
  red[threadIdx.x] = vs;
  __syncthreads();
  for (int st = 64; st > 0; st >>= 1) {
    if (threadIdx.x < st) red[threadIdx.x] += red[threadIdx.x + st];
    __syncthreads();
  }
  float var = red[0] / (float)D_MODEL;
  float inv = 1.0f / sqrtf(var + 1e-6f);
#pragma unroll
  for (int t = 0; t < 4; ++t) {
    int j = threadIdx.x + 128 * t;
    float val = (local[t] - mean) * inv * gamma[j] + beta[j];
    hout[(size_t)r * D_MODEL + j] = val;
    hhout[(size_t)r * D_MODEL + j] = (_Float16)val;
  }
}

// ---------------------------------------------------------------- gathers
__global__ __launch_bounds__(256) void gather_x0_kernel(const float* __restrict__ h,
                                                        float* __restrict__ out0) {
  int idx = blockIdx.x * 256 + threadIdx.x;      // 128*512
  if (idx < NIMG * D_MODEL) {
    int b = idx >> 9, j = idx & 511;
    out0[idx] = h[(size_t)(b * SEQL) * D_MODEL + j];
  }
}

__global__ __launch_bounds__(128) void gather_flat_kernel(const float* __restrict__ h,
                                                          const int* __restrict__ obj_to_img,
                                                          const int* __restrict__ offsets,
                                                          float* __restrict__ flat) {
  int n = blockIdx.x;                            // 8192
  int img = obj_to_img[n];
  int pos = n - offsets[img];
  const float* src = h + (size_t)(img * SEQL + pos + 1) * D_MODEL;
  float* dst = flat + (size_t)n * D_MODEL;
#pragma unroll
  for (int t = 0; t < 4; ++t) {
    int j = threadIdx.x + 128 * t;
    dst[j] = src[j];
  }
}

// ---------------------------------------------------------------- launch

extern "C" void kernel_launch(void* const* d_in, const int* in_sizes, int n_in,
                              void* d_out, int out_size, void* d_ws, size_t ws_size,
                              hipStream_t stream) {
  const float* obj_vecs  = (const float*)d_in[0];
  const float* attr      = (const float*)d_in[1];
  const float* emb_token = (const float*)d_in[2];
  const float* Wq = (const float*)d_in[3];
  const float* bq = (const float*)d_in[4];
  const float* Wk = (const float*)d_in[5];
  const float* bk = (const float*)d_in[6];
  const float* Wv = (const float*)d_in[7];
  const float* bv = (const float*)d_in[8];
  const float* Wo = (const float*)d_in[9];
  const float* bo = (const float*)d_in[10];
  const float* W1 = (const float*)d_in[11];
  const float* b1 = (const float*)d_in[12];
  const float* W2 = (const float*)d_in[13];
  const float* b2 = (const float*)d_in[14];
  const float* ln1_g = (const float*)d_in[15];
  const float* ln1_b = (const float*)d_in[16];
  const float* ln2_g = (const float*)d_in[17];
  const float* ln2_b = (const float*)d_in[18];
  const int*   obj_to_img = (const int*)d_in[19];

  float* out = (float*)d_out;
  float* out_x0    = out;                                   // 128*512
  float* out_masks = out + 65536;                           // 128*128
  float* out_flat  = out + 81920;                           // 8192*512
  float* out_atts  = out + 4276224;                         // 4*128*8*128*128
  const size_t atts_stride = (size_t)NIMG * NHEAD * SEQL * SEQL;  // per layer

  // workspace layout
  const int ROWS = NIMG * SEQL;                             // 16384
  char* ws = (char*)d_ws;
  int*    counts  = (int*)ws;                               ws += 512;
  int*    offsets = (int*)ws;                               ws += 512;
  float*  h   = (float*)ws;                                 ws += (size_t)ROWS * D_MODEL * 4;   // 32 MB
  half_t* hh  = (half_t*)ws;                                ws += (size_t)ROWS * D_MODEL * 2;   // 16 MB
  half_t* qb_ = (half_t*)ws;                                ws += (size_t)ROWS * D_MODEL * 2;   // 16 MB
  half_t* kb_ = (half_t*)ws;                                ws += (size_t)ROWS * D_MODEL * 2;   // 16 MB
  half_t* vb_ = (half_t*)ws;                                ws += (size_t)ROWS * D_MODEL * 2;   // 16 MB
  half_t* ob_ = (half_t*)ws;                                ws += (size_t)ROWS * D_MODEL * 2;   // 16 MB
  half_t* ff1 = (half_t*)ws;                                ws += (size_t)ROWS * DFFN * 2;      // 64 MB
  // per-layer f16 weight staging (reused every layer)
  half_t* wqh = (half_t*)ws;                                ws += (size_t)D_MODEL * D_MODEL * 2;
  half_t* wkh = (half_t*)ws;                                ws += (size_t)D_MODEL * D_MODEL * 2;
  half_t* wvh = (half_t*)ws;                                ws += (size_t)D_MODEL * D_MODEL * 2;
  half_t* woh = (half_t*)ws;                                ws += (size_t)D_MODEL * D_MODEL * 2;
  half_t* w1h = (half_t*)ws;                                ws += (size_t)D_MODEL * DFFN * 2;
  half_t* w2h = (half_t*)ws;                                ws += (size_t)DFFN * D_MODEL * 2;
  float* tbuf = (float*)qb_;   // alias over q+k region (dead when tbuf is live)

  // --- setup
  zero_counts_kernel<<<1, 128, 0, stream>>>(counts);
  count_objs_kernel<<<(NTOT + 255) / 256, 256, 0, stream>>>(obj_to_img, counts);
  scan_offsets_kernel<<<1, 32, 0, stream>>>(counts, offsets);
  write_masks_kernel<<<(NIMG * SEQL + 255) / 256, 256, 0, stream>>>(counts, out_masks);
  init_h_kernel<<<ROWS, 128, 0, stream>>>(emb_token, h, hh);
  fill_objects_kernel<<<NTOT, 128, 0, stream>>>(obj_vecs, attr, obj_to_img, offsets, h, hh);

  const dim3 gemm_d  (D_MODEL / 64, ROWS / 64);             // (8, 256)
  const dim3 gemm_ff (DFFN / 64,    ROWS / 64);             // (32, 256)
  const int NDD  = D_MODEL * D_MODEL;                       // 262144
  const int NDF  = D_MODEL * DFFN;                          // 1048576
  const size_t attn_lds = 64 * 1024 + 32 * 1024;            // 96 KB dynamic

  for (int l = 0; l < NLAYER; ++l) {
    // stage this layer's weights as f16
    f32_to_f16_kernel<<<(NDD + 255) / 256, 256, 0, stream>>>(Wq + (size_t)l * NDD, wqh, NDD);
    f32_to_f16_kernel<<<(NDD + 255) / 256, 256, 0, stream>>>(Wk + (size_t)l * NDD, wkh, NDD);
    f32_to_f16_kernel<<<(NDD + 255) / 256, 256, 0, stream>>>(Wv + (size_t)l * NDD, wvh, NDD);
    f32_to_f16_kernel<<<(NDD + 255) / 256, 256, 0, stream>>>(Wo + (size_t)l * NDD, woh, NDD);
    f32_to_f16_kernel<<<(NDF + 255) / 256, 256, 0, stream>>>(W1 + (size_t)l * NDF, w1h, NDF);
    f32_to_f16_kernel<<<(NDF + 255) / 256, 256, 0, stream>>>(W2 + (size_t)l * NDF, w2h, NDF);

    // Q, K, V projections -> f16
    gemm_wmma_kernel<half_t, false, false><<<gemm_d, 128, 0, stream>>>(
        hh, D_MODEL, wqh, D_MODEL, bq + l * D_MODEL, nullptr, qb_, D_MODEL, D_MODEL);
    gemm_wmma_kernel<half_t, false, false><<<gemm_d, 128, 0, stream>>>(
        hh, D_MODEL, wkh, D_MODEL, bk + l * D_MODEL, nullptr, kb_, D_MODEL, D_MODEL);
    gemm_wmma_kernel<half_t, false, false><<<gemm_d, 128, 0, stream>>>(
        hh, D_MODEL, wvh, D_MODEL, bv + l * D_MODEL, nullptr, vb_, D_MODEL, D_MODEL);

    // attention (writes atts slice + o)
    attn_kernel<<<NIMG * NHEAD, 256, attn_lds, stream>>>(
        qb_, kb_, vb_, out_masks, out_atts + (size_t)l * atts_stride, ob_);

    // o @ Wo + bo + h -> tbuf ; LN1 -> h, hh
    gemm_wmma_kernel<float, false, true><<<gemm_d, 128, 0, stream>>>(
        ob_, D_MODEL, woh, D_MODEL, bo + l * D_MODEL, h, tbuf, D_MODEL, D_MODEL);
    ln_kernel<<<ROWS, 128, 0, stream>>>(tbuf, ln1_g + l * D_MODEL, ln1_b + l * D_MODEL, h, hh);

    // FFN: relu(hh @ W1 + b1) -> ff1 (f16) ; ff1 @ W2 + b2 + h -> tbuf ; LN2 -> h, hh
    gemm_wmma_kernel<half_t, true, false><<<gemm_ff, 128, 0, stream>>>(
        hh, D_MODEL, w1h, DFFN, b1 + l * DFFN, nullptr, ff1, DFFN, D_MODEL);
    gemm_wmma_kernel<float, false, true><<<gemm_d, 128, 0, stream>>>(
        ff1, DFFN, w2h, D_MODEL, b2 + l * D_MODEL, h, tbuf, D_MODEL, DFFN);
    ln_kernel<<<ROWS, 128, 0, stream>>>(tbuf, ln2_g + l * D_MODEL, ln2_b + l * D_MODEL, h, hh);
  }

  gather_x0_kernel<<<(NIMG * D_MODEL + 255) / 256, 256, 0, stream>>>(h, out_x0);
  gather_flat_kernel<<<NTOT, 128, 0, stream>>>(h, obj_to_img, offsets, out_flat);

  (void)in_sizes; (void)n_in; (void)out_size; (void)ws_size;
}
